// SharedChannelPatchConv_35570919145763
// MI455X (gfx1250) — matrix operand order repaired and checked by reading, not compiled, not run on hardware
//
#include <hip/hip_runtime.h>

typedef __attribute__((ext_vector_type(16))) __bf16 v16bf;
typedef __attribute__((ext_vector_type(8)))  float  v8f;

#define BM 128
#define BN 128
#define KSTEPS 8
#define LDSTR 40   // bf16 row stride inside LDS fragment planes (pad for banks, 16B aligned)

__device__ __forceinline__ unsigned f2bf(float f) {
  // round-to-nearest-even f32 -> bf16 bit pattern
  unsigned u = __float_as_uint(f);
  return (u + 0x7FFFu + ((u >> 16) & 1u)) >> 16;
}

__device__ __forceinline__ void split4_store(unsigned short* hiP, unsigned short* loP,
                                             float4 xv) {
  unsigned h0 = f2bf(xv.x), h1 = f2bf(xv.y), h2 = f2bf(xv.z), h3 = f2bf(xv.w);
  unsigned l0 = f2bf(xv.x - __uint_as_float(h0 << 16));
  unsigned l1 = f2bf(xv.y - __uint_as_float(h1 << 16));
  unsigned l2 = f2bf(xv.z - __uint_as_float(h2 << 16));
  unsigned l3 = f2bf(xv.w - __uint_as_float(h3 << 16));
  *reinterpret_cast<uint2*>(hiP) = make_uint2(h0 | (h1 << 16), h2 | (h3 << 16));
  *reinterpret_cast<uint2*>(loP) = make_uint2(l0 | (l1 << 16), l2 | (l3 << 16));
}

union FragCast { uint4 q[2]; v16bf v; };

__device__ __forceinline__ v16bf load_fragA(const unsigned short* plane,
                                            unsigned mrow, unsigned half) {
  // A 16x32 bf16 layout: lane half selects K ranges [half*8, half*8+8) and [16+half*8, ...)
  const unsigned short* base = plane + mrow * LDSTR + half * 8u;
  FragCast u;
  u.q[0] = *reinterpret_cast<const uint4*>(base);
  u.q[1] = *reinterpret_cast<const uint4*>(base + 16);
  return u.v;
}

__device__ __forceinline__ v16bf load_fragB(const unsigned short* plane,
                                            unsigned nrow, unsigned half) {
  // B 32x16 bf16 layout: lane half selects K ranges [half*8, half*8+8) and [16+half*8, ...)
  const unsigned short* base = plane + nrow * LDSTR + half * 8u;
  FragCast u;
  u.q[0] = *reinterpret_cast<const uint4*>(base);
  u.q[1] = *reinterpret_cast<const uint4*>(base + 16);
  return u.v;
}

// ---------------- Stage 1: channel mean (pure bandwidth) ----------------
__global__ __launch_bounds__(256) void mean12_kernel(const float* __restrict__ x,
                                                     float* __restrict__ xm) {
  unsigned t  = blockIdx.x * 256u + threadIdx.x;   // float4 index, 0..1048575
  unsigned b  = t >> 16;                           // 65536 float4 per (b,c) image
  unsigned r4 = t & 65535u;
  const float4* px = reinterpret_cast<const float4*>(x) + (size_t)b * (12u * 65536u) + r4;
  float4 s = make_float4(0.f, 0.f, 0.f, 0.f);
#pragma unroll
  for (int c = 0; c < 12; ++c) {
    float4 v = px[(size_t)c * 65536u];
    s.x += v.x; s.y += v.y; s.z += v.z; s.w += v.w;
  }
  const float inv = 1.0f / 12.0f;
  s.x *= inv; s.y *= inv; s.z *= inv; s.w *= inv;
  reinterpret_cast<float4*>(xm)[t] = s;
}

// ---------------- Stage 2: patch-embed GEMM via bf16x3 WMMA ----------------
// M = 16384 (b*32*32 patches), K = 256 (16x16 patch), N = 768 filters.
__global__ __launch_bounds__(256) void patch_gemm_kernel(
    const float* __restrict__ xm, const float* __restrict__ wgt,
    const float* __restrict__ bias, float* __restrict__ out) {
  __shared__ __align__(16) unsigned short smem[4 * BM * LDSTR];   // 40960 bytes
  unsigned short* aHi = smem;
  unsigned short* aLo = smem + BM * LDSTR;
  unsigned short* bHi = smem + 2 * BM * LDSTR;
  unsigned short* bLo = smem + 3 * BM * LDSTR;

  const unsigned tid  = threadIdx.x;
  const unsigned w    = tid >> 5;         // wave id 0..7
  const unsigned lane = tid & 31u;
  const unsigned half = lane >> 4;
  const unsigned l16  = lane & 15u;
  const unsigned wm   = w >> 1;           // 4 waves along M
  const unsigned wn   = w & 1u;           // 2 waves along N

  const unsigned bx    = blockIdx.x;      // 128 m-blocks x 6 n-blocks = 768
  const unsigned mBase = (bx / 6u) * BM;
  const unsigned oBase = (bx % 6u) * BN;

  v8f acc[2][4];
#pragma unroll
  for (int tm = 0; tm < 2; ++tm)
#pragma unroll
    for (int tn = 0; tn < 4; ++tn)
      acc[tm][tn] = (v8f){};

  for (int ks = 0; ks < KSTEPS; ++ks) {
    // ---- cooperative fill: A patches (f32 -> hi/lo bf16, K-contiguous) ----
#pragma unroll
    for (int r = 0; r < 4; ++r) {
      unsigned q = (unsigned)r * 256u + tid;        // 1024 float4 tasks
      unsigned m = q >> 3, seg = q & 7u;
      unsigned uoff = seg >> 2, v4 = seg & 3u;
      unsigned Mg = mBase + m;
      unsigned b  = Mg >> 10, ij = Mg & 1023u;
      unsigned ii = ij >> 5,  jj = ij & 31u;
      const float4 xv = *reinterpret_cast<const float4*>(
          xm + (size_t)b * 262144u +
          (size_t)(ii * 16u + (unsigned)ks * 2u + uoff) * 512u + jj * 16u + v4 * 4u);
      unsigned kl = uoff * 16u + v4 * 4u;
      split4_store(aHi + m * LDSTR + kl, aLo + m * LDSTR + kl, xv);
    }
    // ---- cooperative fill: B weights ----
#pragma unroll
    for (int r = 0; r < 4; ++r) {
      unsigned q = (unsigned)r * 256u + tid;
      unsigned n = q >> 3, kl = (q & 7u) * 4u;
      unsigned o = oBase + n;
      const float4 wv = *reinterpret_cast<const float4*>(
          wgt + (size_t)o * 256u + (unsigned)ks * 32u + kl);
      split4_store(bHi + n * LDSTR + kl, bLo + n * LDSTR + kl, wv);
    }
    __syncthreads();

    // ---- fragments straight out of LDS (2x ds_load_b128 per half, no unpack) ----
    v16bf ah[2], al[2], bh[4], bl[4];
#pragma unroll
    for (int tm = 0; tm < 2; ++tm) {
      unsigned mrow = wm * 32u + (unsigned)tm * 16u + l16;
      ah[tm] = load_fragA(aHi, mrow, half);
      al[tm] = load_fragA(aLo, mrow, half);
    }
#pragma unroll
    for (int tn = 0; tn < 4; ++tn) {
      unsigned nrow = wn * 64u + (unsigned)tn * 16u + l16;
      bh[tn] = load_fragB(bHi, nrow, half);
      bl[tn] = load_fragB(bLo, nrow, half);
    }
    // ---- bf16x3: hi*hi + hi*lo + lo*hi with f32 accumulation ----
#pragma unroll
    for (int tm = 0; tm < 2; ++tm)
#pragma unroll
      for (int tn = 0; tn < 4; ++tn) {
        acc[tm][tn] = __builtin_amdgcn_wmma_f32_16x16x32_bf16(
            false, ah[tm], false, bh[tn], (short)0, acc[tm][tn], false, false);
        acc[tm][tn] = __builtin_amdgcn_wmma_f32_16x16x32_bf16(
            false, ah[tm], false, bl[tn], (short)0, acc[tm][tn], false, false);
        acc[tm][tn] = __builtin_amdgcn_wmma_f32_16x16x32_bf16(
            false, al[tm], false, bh[tn], (short)0, acc[tm][tn], false, false);
      }
    __syncthreads();
  }

  // ---- epilogue: transpose via LDS so stores are 128B-coalesced along M ----
  float* eps = reinterpret_cast<float*>(smem);
  const unsigned wbase = w * (32u * 33u);            // per-wave region, 2 passes of 32 N
  const unsigned MgL   = mBase + wm * 32u + lane;    // this lane's output row (read phase)
  const unsigned bIdx  = MgL >> 10;
  const unsigned sp    = MgL & 1023u;                // spatial offset i*32+j

#pragma unroll
  for (int p = 0; p < 2; ++p) {
    __syncthreads();
#pragma unroll
    for (int tm = 0; tm < 2; ++tm)
#pragma unroll
      for (int t2 = 0; t2 < 2; ++t2) {
        const v8f c = acc[tm][2 * p + t2];
#pragma unroll
        for (int rr = 0; rr < 8; ++rr)
          eps[wbase + ((unsigned)tm * 16u + half * 8u + (unsigned)rr) * 33u +
              (unsigned)t2 * 16u + l16] = c[rr];
      }
    __syncthreads();
    for (int n2 = 0; n2 < 32; ++n2) {
      unsigned o = oBase + wn * 64u + (unsigned)p * 32u + (unsigned)n2;
      float v = eps[wbase + lane * 33u + (unsigned)n2] + bias[o];
      out[((size_t)(bIdx * 768u + o) << 10) + sp] = v;
    }
  }
}

extern "C" void kernel_launch(void* const* d_in, const int* in_sizes, int n_in,
                              void* d_out, int out_size, void* d_ws, size_t ws_size,
                              hipStream_t stream) {
  const float* x    = (const float*)d_in[0];   // (16,12,512,512) f32
  const float* wgt  = (const float*)d_in[1];   // (768,1,16,16)  f32
  const float* bias = (const float*)d_in[2];   // (768,)         f32
  float* out = (float*)d_out;                  // (16,768,32,32) f32
  float* xm  = (float*)d_ws;                   // 16*512*512 f32 = 16.8 MB scratch

  mean12_kernel<<<4096, 256, 0, stream>>>(x, xm);
  patch_gemm_kernel<<<768, 256, 0, stream>>>(xm, wgt, bias, out);
}
